// IFFT_19490561589807
// MI455X (gfx1250) — matrix operand order repaired and checked by reading, not compiled, not run on hardware
//
#include <hip/hip_runtime.h>
#include <math.h>

// ---------------------------------------------------------------------------
// 1024-point complex IFFT, batch 32768 rows.
//   stage A: 8x 128-point IDFT as matmul  -> bf16 split-precision WMMA
//   stage B: 3 radix-2 butterfly stages   -> fp32 VALU in LDS
// Data movement: GLOBAL_LOAD_ASYNC_TO_LDS / GLOBAL_STORE_ASYNC_FROM_LDS
// ---------------------------------------------------------------------------

typedef __attribute__((ext_vector_type(16))) __bf16 v16bf;
typedef __attribute__((ext_vector_type(8)))  __bf16 v8bf;
typedef __attribute__((ext_vector_type(8)))  float  v8f;

#define N_FFT       1024
#define NSUB        128
#define ROWS_PER_WG 16
#define XSTRIDE     1028   // staged-row stride in floats (pad 4 -> bank spread)
#define BSTRIDE     136    // B-matrix row stride in bf16 (pad 8 -> bank spread)

static __device__ inline v16bf ldsB(const __bf16* p) {
  // 32B fragment as two aligned 16B LDS loads
  v8bf a = *(const v8bf*)p;
  v8bf b = *(const v8bf*)(p + 8);
  return __builtin_shufflevector(a, b, 0,1,2,3,4,5,6,7,8,9,10,11,12,13,14,15);
}

// Async DMA: global -> LDS, 16 bytes per lane. LDS destination is the byte
// offset (low 32 bits of the flat pointer per the LDS aperture truncation).
static __device__ inline void async_ld16(unsigned int lds_off, const void* gaddr) {
  asm volatile("global_load_async_to_lds_b128 %0, %1, off"
               :: "v"(lds_off), "v"(gaddr) : "memory");
}
// Async DMA: LDS -> global, 16 bytes per lane.
static __device__ inline void async_st16(void* gaddr, unsigned int lds_off) {
  asm volatile("global_store_async_from_lds_b128 %0, %1, off"
               :: "v"(gaddr), "v"(lds_off) : "memory");
}
static __device__ inline void wait_async0() {
  asm volatile("s_wait_asynccnt 0x0" ::: "memory");
}

#define WMMA_BF16(A,B,C) \
  __builtin_amdgcn_wmma_f32_16x16x32_bf16(false,(A),false,(B),(short)0,(C),false,false)

__global__ __launch_bounds__(256, 1)
void ifft1024_kernel(const float* __restrict__ re, const float* __restrict__ im,
                     float* __restrict__ out, int nrows) {
  extern __shared__ char smem[];
  float*  sXr = (float*)smem;                        // 16 x 1028 f32
  float*  sXi = sXr + ROWS_PER_WG * XSTRIDE;         // 16 x 1028 f32
  __bf16* sRh = (__bf16*)(sXi + ROWS_PER_WG * XSTRIDE);
  __bf16* sRl = sRh + NSUB * BSTRIDE;
  __bf16* sIh = sRl + NSUB * BSTRIDE;
  __bf16* sIl = sIh + NSUB * BSTRIDE;

  const int tid  = threadIdx.x;
  const int lane = tid & 31;
  const int wv   = tid >> 5;                         // wave id == chunk id 0..7
  const int row0 = blockIdx.x * ROWS_PER_WG;

  // ---- Phase 1 (issue first): async DMA of 16 rows re/im into LDS ----------
#pragma unroll
  for (int m = 0; m < ROWS_PER_WG; ++m) {
    const int row = row0 + m;
    if (row < nrows) {
      async_ld16((unsigned int)(size_t)(sXr + m * XSTRIDE) + (unsigned int)tid * 16u,
                 re + (size_t)row * N_FFT + tid * 4);
      async_ld16((unsigned int)(size_t)(sXi + m * XSTRIDE) + (unsigned int)tid * 16u,
                 im + (size_t)row * N_FFT + tid * 4);
    } else {
      ((float4*)(sXr + m * XSTRIDE))[tid] = make_float4(0.f, 0.f, 0.f, 0.f);
      ((float4*)(sXi + m * XSTRIDE))[tid] = make_float4(0.f, 0.f, 0.f, 0.f);
    }
  }

  // ---- Phase 0 (overlapped with DMA): IDFT128 matrices as bf16 hi/lo -------
  for (int idx = tid; idx < NSUB * NSUB; idx += 256) {
    const int n = idx >> 7, k = idx & 127;
    const float ang = (float)((n * k) & 127) * 0.049087385212340517f; // 2pi/128
    float s, c;
    __sincosf(ang, &s, &c);
    const float rr = c * 0.0078125f, ri = s * 0.0078125f;             // 1/128
    const __bf16 rh = (__bf16)rr, ih = (__bf16)ri;
    sRh[n * BSTRIDE + k] = rh;  sRl[n * BSTRIDE + k] = (__bf16)(rr - (float)rh);
    sIh[n * BSTRIDE + k] = ih;  sIl[n * BSTRIDE + k] = (__bf16)(ri - (float)ih);
  }
  wait_async0();
  __syncthreads();

  // ---- Phase 2: per-wave chunk IDFT via split-bf16 WMMA --------------------
  // chunk c reads permuted column k*8 + bitrev3(c)
  const int rc = ((wv & 1) << 2) | (wv & 2) | ((wv & 4) >> 2);

  // A fragments (16x32 bf16 layout): lanes 0-15 row M=lane, K 0..15;
  // lanes 16-31 row M=lane-16, K 16..31
  v16bf aRh[4], aRl[4], aIh[4], aIl[4];
  {
    const int m  = lane & 15;
    const int kb = (lane >> 4) << 4;                 // 0 or 16
    const float* pr = sXr + m * XSTRIDE + rc;
    const float* pi = sXi + m * XSTRIDE + rc;
#pragma unroll
    for (int kt = 0; kt < 4; ++kt) {
#pragma unroll
      for (int e = 0; e < 16; ++e) {
        const int K = kt * 32 + kb + e;
        const float fr = pr[K << 3];                 // permutation stride 8
        const float fi = pi[K << 3];
        __bf16 h = (__bf16)fr;
        aRh[kt][e] = h;  aRl[kt][e] = (__bf16)(fr - (float)h);
        h = (__bf16)fi;
        aIh[kt][e] = h;  aIl[kt][e] = (__bf16)(fi - (float)h);
      }
    }
  }
  __syncthreads();   // everyone captured A before D tiles overwrite staging

  const int colBase = wv * NSUB;
  for (int nt = 0; nt < 8; ++nt) {
    v8f accR = {};   // re@R
    v8f accN = {};   // im@I   (subtracted at store -> no B negation, no hazards)
    v8f accI = {};   // re@I + im@R
    const int n   = nt * 16 + (lane & 15);
    const int khi = (lane >> 4) << 4;                // B: lanes 0-15 K 0..15, 16-31 K 16..31
#pragma unroll
    for (int kt = 0; kt < 4; ++kt) {
      const int kb = kt * 32 + khi;
      const v16bf brh = ldsB(sRh + n * BSTRIDE + kb);
      const v16bf brl = ldsB(sRl + n * BSTRIDE + kb);
      const v16bf bih = ldsB(sIh + n * BSTRIDE + kb);
      const v16bf bil = ldsB(sIl + n * BSTRIDE + kb);
      // 3-term split-precision products
      accR = WMMA_BF16(aRh[kt], brh, accR);
      accR = WMMA_BF16(aRl[kt], brh, accR);
      accR = WMMA_BF16(aRh[kt], brl, accR);
      accN = WMMA_BF16(aIh[kt], bih, accN);
      accN = WMMA_BF16(aIl[kt], bih, accN);
      accN = WMMA_BF16(aIh[kt], bil, accN);
      accI = WMMA_BF16(aRh[kt], bih, accI);
      accI = WMMA_BF16(aRl[kt], bih, accI);
      accI = WMMA_BF16(aRh[kt], bil, accI);
      accI = WMMA_BF16(aIh[kt], brh, accI);
      accI = WMMA_BF16(aIl[kt], brh, accI);
      accI = WMMA_BF16(aIh[kt], brl, accI);
    }
    // D tile: lanes 0-15 -> M 0..7 (VGPR v), lanes 16-31 -> M 8..15
    const int col = colBase + nt * 16 + (lane & 15);
    const int mb  = (lane >> 4) << 3;
#pragma unroll
    for (int v = 0; v < 8; ++v) {
      sXr[(mb + v) * XSTRIDE + col] = accR[v] - accN[v];   // yr = re@R - im@I
      sXi[(mb + v) * XSTRIDE + col] = accI[v];             // yi = re@I + im@R
    }
  }
  __syncthreads();

  // ---- Phase 3: three radix-2 butterfly stages in LDS (fp32) ---------------
  // 1/8 IFFT normalizer folded into the last stage so LDS holds final output.
#pragma unroll
  for (int s = 0; s < 3; ++s) {
    const int   half  = NSUB << s;                   // 128, 256, 512
    const float wstep = 3.14159265358979323846f / (float)half; // 2pi/(2*half)
    const float sc    = (s == 2) ? 0.125f : 1.0f;
    for (int w = tid; w < ROWS_PER_WG * 512; w += 256) {
      const int m = w >> 9;
      const int r = w & 511;
      const int g = r >> (7 + s);
      const int t = r & (half - 1);
      const int i0 = m * XSTRIDE + g * (half << 1) + t;
      const int i1 = i0 + half;
      float si, co;
      __sincosf(wstep * (float)t, &si, &co);
      const float er  = sXr[i0], ei  = sXi[i0];
      const float odr = sXr[i1], odi = sXi[i1];
      const float tr = co * odr - si * odi;
      const float ti = co * odi + si * odr;
      sXr[i0] = (er + tr) * sc;  sXi[i0] = (ei + ti) * sc;
      sXr[i1] = (er - tr) * sc;  sXi[i1] = (ei - ti) * sc;
    }
    __syncthreads();
  }

  // ---- Phase 4: async store straight from LDS (yr block, then yi block) ----
  float* outI = out + (size_t)nrows * N_FFT;
#pragma unroll
  for (int m = 0; m < ROWS_PER_WG; ++m) {
    const int row = row0 + m;
    if (row < nrows) {
      async_st16(out  + (size_t)row * N_FFT + tid * 4,
                 (unsigned int)(size_t)(sXr + m * XSTRIDE) + (unsigned int)tid * 16u);
      async_st16(outI + (size_t)row * N_FFT + tid * 4,
                 (unsigned int)(size_t)(sXi + m * XSTRIDE) + (unsigned int)tid * 16u);
    }
  }
  wait_async0();
}

extern "C" void kernel_launch(void* const* d_in, const int* in_sizes, int n_in,
                              void* d_out, int out_size, void* d_ws, size_t ws_size,
                              hipStream_t stream) {
  (void)n_in; (void)out_size; (void)d_ws; (void)ws_size;
  const float* re = (const float*)d_in[0];
  const float* im = (const float*)d_in[1];
  float* out = (float*)d_out;

  const int nrows   = in_sizes[0] / N_FFT;           // 8*4096 = 32768
  const int nblocks = (nrows + ROWS_PER_WG - 1) / ROWS_PER_WG;

  const size_t smem = (size_t)(2 * ROWS_PER_WG * XSTRIDE) * sizeof(float)
                    + (size_t)(4 * NSUB * BSTRIDE) * sizeof(__bf16);

  hipFuncSetAttribute((const void*)ifft1024_kernel,
                      hipFuncAttributeMaxDynamicSharedMemorySize, (int)smem);

  ifft1024_kernel<<<nblocks, 256, smem, stream>>>(re, im, out, nrows);
}
